// LiquidNeuralNetwork_1005022347994
// MI455X (gfx1250) — compile-verified, matrix operand order checked by reading, not run on hardware
//
#include <hip/hip_runtime.h>

// ---------------- problem constants ----------------
constexpr int Bc = 4, Sc = 2048, Vc = 8192, Ec = 256, Hc = 256, NHc = 8, HDc = 32;
constexpr int Mc = Bc * Sc;               // 8192 token rows
constexpr int CL = 128, NC = Sc / CL;     // scan chunking: 16 chunks of 128

// ---------------- WMMA types / helpers ----------------
typedef __attribute__((ext_vector_type(16))) __bf16 v16bf;
typedef __attribute__((ext_vector_type(8)))  float  v8f;
typedef __attribute__((ext_vector_type(4)))  unsigned int v4u;
typedef __attribute__((ext_vector_type(8)))  int          v8i;
typedef __attribute__((ext_vector_type(4)))  int          v4i;

#if __has_builtin(__builtin_amdgcn_tensor_load_to_lds) && __has_builtin(__builtin_amdgcn_s_wait_tensorcnt)
#define HAS_TDM 1
#endif

union FragBF {
    v16bf v;
    uint4 q[2];
    unsigned short h[16];
};

__device__ __forceinline__ unsigned short f2bf(float f) {
    union { float f; unsigned u; } x{f};
    unsigned r = x.u + 0x7FFFu + ((x.u >> 16) & 1u);   // round-to-nearest-even
    return (unsigned short)(r >> 16);
}

// A-matrix 16x32 bf16 fragment (also valid for B-from-[N,K]-row-major, per ISA 7.12.2):
// lane%16 = row (M or N), lane/16 picks K-octet; vgprs 0-3 hold K=bk..bk+7, vgprs 4-7 K=bk+16..bk+23.
__device__ __forceinline__ FragBF load_frag(const unsigned short* base, int ld, int lane) {
    int mn = lane & 15;
    int bk = (lane >> 4) << 3;
    const unsigned short* p = base + mn * ld + bk;
    FragBF f;
    f.q[0] = *reinterpret_cast<const uint4*>(p);
    f.q[1] = *reinterpret_cast<const uint4*>(p + 16);
    return f;
}

__device__ __forceinline__ v8f wmma_bf16(const FragBF& a, const FragBF& b, v8f c) {
    return __builtin_amdgcn_wmma_f32_16x16x32_bf16(false, a.v, false, b.v, (short)0, c, false, false);
}

#ifdef HAS_TDM
// Low 32 bits of a generic pointer to LDS = LDS byte offset (ISA aperture mapping)
__device__ __forceinline__ unsigned lds_off(const void* p) {
    return (unsigned)(unsigned long long)(uintptr_t)p;
}

// Issue a 2-D TDM tile load: tile_d0 elems/row (bf16), tile_d1 rows,
// row stride = stride_e elements, packed row-major into LDS at lds_byte.
__device__ __forceinline__ void tdm_load_2d(unsigned lds_byte, const void* gptr,
                                            unsigned tile_d0, unsigned tile_d1,
                                            unsigned long long stride_e) {
    unsigned long long ga = (unsigned long long)(uintptr_t)gptr;
    v4u g0;
    g0[0] = 1u;                                              // count=1, user mode
    g0[1] = lds_byte;                                        // lds_addr
    g0[2] = (unsigned)(ga & 0xFFFFFFFFu);                    // global_addr[31:0]
    g0[3] = (unsigned)((ga >> 32) & 0x01FFFFFFu) | (2u << 30); // addr[56:32] | type=2
    const unsigned td0 = 1u << 30, td1 = 1u << 30;           // huge tensor dims (in-bounds by construction)
    v8i g1;
    g1[0] = (int)(1u << 16);                                 // data_size=1 -> 2 bytes
    g1[1] = (int)((td0 & 0xFFFFu) << 16);                    // tensor_dim0[15:0] @bit48
    g1[2] = (int)(((td0 >> 16) & 0xFFFFu) | ((td1 & 0xFFFFu) << 16));
    g1[3] = (int)(((td1 >> 16) & 0xFFFFu) | ((tile_d0 & 0xFFFFu) << 16));
    g1[4] = (int)(tile_d1 & 0xFFFFu);                        // tile_dim1 (tile_dim2=0)
    g1[5] = (int)(unsigned)(stride_e & 0xFFFFFFFFull);       // tensor_dim0_stride lo
    g1[6] = (int)(unsigned)((stride_e >> 32) & 0xFFFFull);   // stride hi16 (dim1_stride=0)
    g1[7] = 0;
    v4i z4 = {0, 0, 0, 0};
    v8i z8 = {0, 0, 0, 0, 0, 0, 0, 0};
    // 6-arg variant (clang-23 / therock-10.0 headers): groups 0..3 + extra group + cpol
    __builtin_amdgcn_tensor_load_to_lds(g0, g1, z4, z4, z8, 0);
}
#endif

// ---------------- small utility kernels ----------------
__global__ __launch_bounds__(256) void conv_f32_bf16(const float* __restrict__ src,
                                                     unsigned short* __restrict__ dst, int n) {
    int i = blockIdx.x * 256 + threadIdx.x;
    if (i < n) dst[i] = f2bf(src[i]);
}

__global__ __launch_bounds__(256) void embed_kernel(const int* __restrict__ tokens,
                                                    const float* __restrict__ emb,
                                                    unsigned short* __restrict__ xb) {
    int row = blockIdx.x;
    int e   = threadIdx.x;
    int tok = tokens[row];
    xb[(size_t)row * Ec + e] = f2bf(emb[(size_t)tok * Ec + e]);
}

// ---------------- bf16 WMMA GEMM: C[M,N] = A[M,K] * W[N,K]^T + bias ----------------
// MODE 0: f32 out, 1: bf16 out (val*scale), 2: bf16 scatter to [B,NH,S,HD] (val*scale)
template <int MODE>
__global__ __launch_bounds__(256) void gemm_bf16_kernel(const unsigned short* __restrict__ A,
                                                        const unsigned short* __restrict__ W,
                                                        const float* __restrict__ bias,
                                                        void* __restrict__ Cout,
                                                        int M, int N, int K, float scale) {
    constexpr int BM = 128, BN = 128, BK = 32;
    __shared__ __align__(16) unsigned short Bs[2][BN * BK];   // double buffer (TDM path)

    int tid  = threadIdx.x;
    int lane = tid & 31, w = tid >> 5;
    int mTile = blockIdx.x * BM, nTile = blockIdx.y * BN;

    v8f zero = {0.f, 0.f, 0.f, 0.f, 0.f, 0.f, 0.f, 0.f};
    v8f acc[8];
#pragma unroll
    for (int t = 0; t < 8; ++t) acc[t] = zero;

    int mRow = mTile + (w << 4) + (lane & 15);
    int bk   = (lane >> 4) << 3;
    const unsigned short* aPtr = A + (size_t)mRow * K + bk;
    const int nk = K / BK;

#ifdef HAS_TDM
    // ---- double-buffered Tensor Data Mover pipeline for the shared B tile ----
    if (w == 0)
        tdm_load_2d(lds_off(&Bs[0][0]), W + (size_t)nTile * K, BK, BN, (unsigned long long)K);
    int cur = 0;
    for (int i = 0; i < nk; ++i) {
        int kb = i * BK;
        if (w == 0) {
            if (i + 1 < nk) {
                tdm_load_2d(lds_off(&Bs[cur ^ 1][0]), W + (size_t)nTile * K + kb + BK,
                            BK, BN, (unsigned long long)K);
                __builtin_amdgcn_s_wait_tensorcnt((short)1);   // tile i complete
            } else {
                __builtin_amdgcn_s_wait_tensorcnt((short)0);
            }
        }
        __syncthreads();

        FragBF aF;
        aF.q[0] = *reinterpret_cast<const uint4*>(aPtr + kb);
        aF.q[1] = *reinterpret_cast<const uint4*>(aPtr + kb + 16);
        if (kb + BK < K) __builtin_prefetch(aPtr + kb + BK, 0, 1);

#pragma unroll
        for (int t = 0; t < 8; ++t) {
            FragBF bF = load_frag(&Bs[cur][(t * 16) * BK], BK, lane);
            acc[t] = wmma_bf16(aF, bF, acc[t]);
        }
        __syncthreads();
        cur ^= 1;
    }
#else
    // ---- manual cooperative staging fallback ----
    int nloc = tid >> 1, half = tid & 1;
    for (int kb = 0; kb < K; kb += BK) {
        const unsigned short* bsrc = W + (size_t)(nTile + nloc) * K + kb + (half << 4);
        *reinterpret_cast<uint4*>(&Bs[0][nloc * BK + (half << 4)]) =
            *reinterpret_cast<const uint4*>(bsrc);
        __syncthreads();
        FragBF aF;
        aF.q[0] = *reinterpret_cast<const uint4*>(aPtr + kb);
        aF.q[1] = *reinterpret_cast<const uint4*>(aPtr + kb + 16);
        if (kb + BK < K) __builtin_prefetch(aPtr + kb + BK, 0, 1);
#pragma unroll
        for (int t = 0; t < 8; ++t) {
            FragBF bF = load_frag(&Bs[0][(t * 16) * BK], BK, lane);
            acc[t] = wmma_bf16(aF, bF, acc[t]);
        }
        __syncthreads();
    }
#endif

    // epilogue — C/D layout: vgpr r -> row r + 8*(lane>>4); col = lane&15
    int cl = lane & 15, hi = lane >> 4;
#pragma unroll
    for (int t = 0; t < 8; ++t) {
        int nCol = nTile + t * 16 + cl;
        float bv = bias[nCol];
#pragma unroll
        for (int r = 0; r < 8; ++r) {
            int m = mTile + (w << 4) + r + (hi << 3);
            float val = acc[t][r] + bv;
            if (MODE == 0) {
                ((float*)Cout)[(size_t)m * N + nCol] = val;
            } else if (MODE == 1) {
                ((unsigned short*)Cout)[(size_t)m * N + nCol] = f2bf(val * scale);
            } else {
                val *= scale;
                int bb = m / Sc, ss = m % Sc;
                int nh = nCol / HDc, d = nCol % HDc;
                ((unsigned short*)Cout)[(((size_t)bb * NHc + nh) * Sc + ss) * HDc + d] = f2bf(val);
            }
        }
    }
}

// ---------------- liquid scan (3-pass chunked parallel scan) ----------------
__global__ __launch_bounds__(256) void scan_pass1(const float* __restrict__ I,
                                                  const float* __restrict__ tau,
                                                  const float* __restrict__ log_dt,
                                                  float* __restrict__ cU, float* __restrict__ cS) {
    int h = threadIdx.x, c = blockIdx.x, b = blockIdx.y;
    float a = __expf(log_dt[0]) / tau[h];
    float u = 0.f, s = 0.f;
    const float* ip = I + ((size_t)b * Sc + c * CL) * Hc + h;
    for (int t = 0; t < CL; ++t) {
        float It = ip[(size_t)t * Hc];
        float sn = s + a * (u - s);
        u = u + a * (It - u);
        s = sn;
    }
    size_t idx = ((size_t)b * Hc + h) * NC + c;
    cU[idx] = u;
    cS[idx] = s;
}

__global__ __launch_bounds__(256) void scan_pass2(const float* __restrict__ cU,
                                                  const float* __restrict__ cS,
                                                  float* __restrict__ uIn, float* __restrict__ sIn,
                                                  const float* __restrict__ tau,
                                                  const float* __restrict__ log_dt) {
    int idx = blockIdx.x * 256 + threadIdx.x;     // b*H + h
    int h = idx & (Hc - 1);
    float a = __expf(log_dt[0]) / tau[h];
    float r = 1.f - a;
    float rp = 1.f, bpow = r;                     // r^(CL-1) by squaring
    int e = CL - 1;
    while (e) { if (e & 1) rp *= bpow; bpow *= bpow; e >>= 1; }
    float d  = rp * r;                            // r^CL
    float cc = (float)CL * a * rp;                // CL * a * r^(CL-1)
    float u = 0.f, s = 0.f;
    for (int c = 0; c < NC; ++c) {
        size_t o = (size_t)idx * NC + c;
        uIn[o] = u;  sIn[o] = s;
        float vU = cU[o], vS = cS[o];
        float un = d * u + vU;
        float sn = cc * u + d * s + vS;
        u = un; s = sn;
    }
}

__global__ __launch_bounds__(256) void scan_pass3(const float* __restrict__ I,
                                                  const float* __restrict__ tau,
                                                  const float* __restrict__ log_dt,
                                                  const float* __restrict__ uIn,
                                                  const float* __restrict__ sIn,
                                                  unsigned short* __restrict__ xout) {
    int h = threadIdx.x, c = blockIdx.x, b = blockIdx.y;
    float a = __expf(log_dt[0]) / tau[h];
    size_t sidx = ((size_t)b * Hc + h) * NC + c;
    float u = uIn[sidx], s = sIn[sidx];
    const float* ip = I + ((size_t)b * Sc + c * CL) * Hc + h;
    unsigned short* op = xout + ((size_t)b * Sc + c * CL) * Hc + h;
    for (int t = 0; t < CL; ++t) {
        float It = ip[(size_t)t * Hc];
        float sn = s + a * (u - s);
        u = u + a * (It - u);
        s = sn;
        op[(size_t)t * Hc] = f2bf(tanhf(s));
    }
}

// ---------------- flash attention, TK=64 keys/iter (q pre-scaled by 1/sqrt(HD)) ----------------
// q,k,v laid out [B,NH,S,HD] bf16; ctx written [B,S,H] bf16
__global__ __launch_bounds__(256) void flash_attn_kernel(const unsigned short* __restrict__ Q,
                                                         const unsigned short* __restrict__ Kh,
                                                         const unsigned short* __restrict__ Vh,
                                                         unsigned short* __restrict__ ctx) {
    constexpr int TK = 64;
    __shared__ __align__(16) unsigned short Ks[TK * 32];       // [key][hd]
    __shared__ __align__(16) unsigned short Vt[32 * TK];       // [hd][key] (transposed)
    __shared__ __align__(16) unsigned short Ps[8 * 16 * TK];   // per-wave P tile

    int tid = threadIdx.x, lane = tid & 31, w = tid >> 5;
    int bh = blockIdx.y;
    int b = bh >> 3, nh = bh & 7;
    int q0 = blockIdx.x * 128 + (w << 4);

    const unsigned short* qh = Q + (size_t)bh * Sc * HDc;
    const unsigned short* kp = Kh + (size_t)bh * Sc * HDc;
    const unsigned short* vp = Vh + (size_t)bh * Sc * HDc;

    FragBF aQ = load_frag(qh + (size_t)q0 * HDc, HDc, lane);

    v8f zero = {0.f, 0.f, 0.f, 0.f, 0.f, 0.f, 0.f, 0.f};
    v8f O0 = zero, O1 = zero;
    float rowM[8], rowL[8];
#pragma unroll
    for (int r = 0; r < 8; ++r) { rowM[r] = -3.0e38f; rowL[r] = 0.f; }

    unsigned short* Pw = &Ps[w * 16 * TK];
    int n0 = lane & 15, hi = lane >> 4;

    for (int kb = 0; kb < Sc; kb += TK) {
        // ---- staging: K tile via TDM (contiguous 4KB), V tile transposed manually ----
#ifdef HAS_TDM
        if (w == 0) {
            tdm_load_2d(lds_off(&Ks[0]), kp + (size_t)kb * HDc, TK * 32, 1, TK * 32ull);
            __builtin_amdgcn_s_wait_tensorcnt((short)0);
        }
#else
        {
            int e = tid * 8;
            *reinterpret_cast<uint4*>(&Ks[e]) =
                *reinterpret_cast<const uint4*>(&kp[(size_t)kb * HDc + e]);
        }
#endif
        {
            int e = tid * 8;                                 // 256 threads * 8 = 2048 elems
            uint4 vv = *reinterpret_cast<const uint4*>(&vp[(size_t)kb * HDc + e]);
            const unsigned short* vs = reinterpret_cast<const unsigned short*>(&vv);
            int key = e >> 5, hd = e & 31;                   // 8 consecutive hd, same key
#pragma unroll
            for (int i = 0; i < 8; ++i) Vt[(hd + i) * TK + key] = vs[i];
        }
        __syncthreads();

        FragBF bK0 = load_frag(&Ks[0 * 16 * 32], 32, lane);
        FragBF bK1 = load_frag(&Ks[1 * 16 * 32], 32, lane);
        FragBF bK2 = load_frag(&Ks[2 * 16 * 32], 32, lane);
        FragBF bK3 = load_frag(&Ks[3 * 16 * 32], 32, lane);

        v8f sc0 = wmma_bf16(aQ, bK0, zero);
        v8f sc1 = wmma_bf16(aQ, bK1, zero);
        v8f sc2 = wmma_bf16(aQ, bK2, zero);
        v8f sc3 = wmma_bf16(aQ, bK3, zero);

#pragma unroll
        for (int r = 0; r < 8; ++r) {
            float s0 = sc0[r], s1 = sc1[r], s2 = sc2[r], s3 = sc3[r];
            float mx = fmaxf(fmaxf(s0, s1), fmaxf(s2, s3));
            mx = fmaxf(mx, __shfl_xor(mx, 1, 32));
            mx = fmaxf(mx, __shfl_xor(mx, 2, 32));
            mx = fmaxf(mx, __shfl_xor(mx, 4, 32));
            mx = fmaxf(mx, __shfl_xor(mx, 8, 32));
            float nm   = fmaxf(rowM[r], mx);
            float corr = __expf(rowM[r] - nm);
            float p0 = __expf(s0 - nm), p1 = __expf(s1 - nm);
            float p2 = __expf(s2 - nm), p3 = __expf(s3 - nm);
            float rs = (p0 + p1) + (p2 + p3);
            rs += __shfl_xor(rs, 1, 32);
            rs += __shfl_xor(rs, 2, 32);
            rs += __shfl_xor(rs, 4, 32);
            rs += __shfl_xor(rs, 8, 32);
            rowL[r] = rowL[r] * corr + rs;
            rowM[r] = nm;
            O0[r] *= corr;
            O1[r] *= corr;
            int m = r + (hi << 3);
            Pw[m * TK +  0 + n0] = f2bf(p0);
            Pw[m * TK + 16 + n0] = f2bf(p1);
            Pw[m * TK + 32 + n0] = f2bf(p2);
            Pw[m * TK + 48 + n0] = f2bf(p3);
        }

        // P (C-layout) -> A-layout via wave-private LDS round-trip; K-dim 64 = 2 frags
        FragBF aP0 = load_frag(Pw +  0, TK, lane);
        FragBF aP1 = load_frag(Pw + 32, TK, lane);
        FragBF bV00 = load_frag(&Vt[0 * 16 * TK +  0], TK, lane);
        FragBF bV01 = load_frag(&Vt[0 * 16 * TK + 32], TK, lane);
        FragBF bV10 = load_frag(&Vt[1 * 16 * TK +  0], TK, lane);
        FragBF bV11 = load_frag(&Vt[1 * 16 * TK + 32], TK, lane);
        O0 = wmma_bf16(aP0, bV00, O0);
        O0 = wmma_bf16(aP1, bV01, O0);
        O1 = wmma_bf16(aP0, bV10, O1);
        O1 = wmma_bf16(aP1, bV11, O1);
        __syncthreads();
    }

    // ---- normalize and scatter to [B,S,H] ----
#pragma unroll
    for (int r = 0; r < 8; ++r) {
        float inv = 1.f / rowL[r];
        int m = r + (hi << 3);
        size_t srow = (size_t)b * Sc + q0 + m;
        ctx[srow * Hc + nh * HDc + n0]      = f2bf(O0[r] * inv);
        ctx[srow * Hc + nh * HDc + 16 + n0] = f2bf(O1[r] * inv);
    }
}

// ---------------- workspace layout (bytes) ----------------
constexpr size_t OFF_XB  = 0;
constexpr size_t OFF_I   = OFF_XB  + (size_t)Mc * Hc * 2;
constexpr size_t OFF_XB2 = OFF_I   + (size_t)Mc * Hc * 4;
constexpr size_t OFF_Q   = OFF_XB2 + (size_t)Mc * Hc * 2;
constexpr size_t OFF_K   = OFF_Q   + (size_t)Mc * Hc * 2;
constexpr size_t OFF_V   = OFF_K   + (size_t)Mc * Hc * 2;
constexpr size_t OFF_CTX = OFF_V   + (size_t)Mc * Hc * 2;
constexpr size_t OFF_WL  = OFF_CTX + (size_t)Mc * Hc * 2;
constexpr size_t OFF_WQ  = OFF_WL  + (size_t)2 * Hc * Ec * 2;
constexpr size_t OFF_WK  = OFF_WQ  + (size_t)Hc * Hc * 2;
constexpr size_t OFF_WV  = OFF_WK  + (size_t)Hc * Hc * 2;
constexpr size_t OFF_WO  = OFF_WV  + (size_t)Hc * Hc * 2;
constexpr size_t OFF_WH  = OFF_WO  + (size_t)Hc * Hc * 2;
constexpr size_t OFF_CU  = OFF_WH  + (size_t)Vc * Hc * 2;
constexpr size_t OFF_CS  = OFF_CU  + (size_t)Bc * Hc * NC * 4;
constexpr size_t OFF_UIN = OFF_CS  + (size_t)Bc * Hc * NC * 4;
constexpr size_t OFF_SIN = OFF_UIN + (size_t)Bc * Hc * NC * 4;

extern "C" void kernel_launch(void* const* d_in, const int* in_sizes, int n_in,
                              void* d_out, int out_size, void* d_ws, size_t ws_size,
                              hipStream_t stream) {
    const int*   tokens = (const int*)  d_in[0];
    const float* emb    = (const float*)d_in[1];
    const float* Ws     = (const float*)d_in[2];
    const float* bs     = (const float*)d_in[3];
    const float* taus   = (const float*)d_in[4];
    const float* qw     = (const float*)d_in[5];
    const float* qb     = (const float*)d_in[6];
    const float* kw     = (const float*)d_in[7];
    const float* kb     = (const float*)d_in[8];
    const float* vw     = (const float*)d_in[9];
    const float* vb     = (const float*)d_in[10];
    const float* ow     = (const float*)d_in[11];
    const float* ob     = (const float*)d_in[12];
    const float* head_w = (const float*)d_in[13];
    const float* head_b = (const float*)d_in[14];
    const float* log_dt = (const float*)d_in[15];

    char* ws = (char*)d_ws;
    unsigned short* XB  = (unsigned short*)(ws + OFF_XB);
    float*          I   = (float*)         (ws + OFF_I);
    unsigned short* XB2 = (unsigned short*)(ws + OFF_XB2);
    unsigned short* Qb  = (unsigned short*)(ws + OFF_Q);
    unsigned short* Kb  = (unsigned short*)(ws + OFF_K);
    unsigned short* Vb  = (unsigned short*)(ws + OFF_V);
    unsigned short* CTX = (unsigned short*)(ws + OFF_CTX);
    unsigned short* WL  = (unsigned short*)(ws + OFF_WL);
    unsigned short* WQ  = (unsigned short*)(ws + OFF_WQ);
    unsigned short* WK  = (unsigned short*)(ws + OFF_WK);
    unsigned short* WV  = (unsigned short*)(ws + OFF_WV);
    unsigned short* WO  = (unsigned short*)(ws + OFF_WO);
    unsigned short* WH  = (unsigned short*)(ws + OFF_WH);
    float* CU  = (float*)(ws + OFF_CU);
    float* CS  = (float*)(ws + OFF_CS);
    float* UIN = (float*)(ws + OFF_UIN);
    float* SIN = (float*)(ws + OFF_SIN);

    conv_f32_bf16<<<(2 * Hc * Ec + 255) / 256, 256, 0, stream>>>(Ws, WL, 2 * Hc * Ec);
    conv_f32_bf16<<<(Hc * Hc + 255) / 256, 256, 0, stream>>>(qw, WQ, Hc * Hc);
    conv_f32_bf16<<<(Hc * Hc + 255) / 256, 256, 0, stream>>>(kw, WK, Hc * Hc);
    conv_f32_bf16<<<(Hc * Hc + 255) / 256, 256, 0, stream>>>(vw, WV, Hc * Hc);
    conv_f32_bf16<<<(Hc * Hc + 255) / 256, 256, 0, stream>>>(ow, WO, Hc * Hc);
    conv_f32_bf16<<<(Vc * Hc + 255) / 256, 256, 0, stream>>>(head_w, WH, Vc * Hc);

    embed_kernel<<<Mc, 256, 0, stream>>>(tokens, emb, XB);

    for (int l = 0; l < 2; ++l) {
        gemm_bf16_kernel<0><<<dim3(Mc / 128, Hc / 128), 256, 0, stream>>>(
            XB, WL + (size_t)l * Hc * Ec, bs + (size_t)l * Hc, I, Mc, Hc, Ec, 1.0f);
        scan_pass1<<<dim3(NC, Bc), 256, 0, stream>>>(I, taus + (size_t)l * Hc, log_dt, CU, CS);
        scan_pass2<<<(Bc * Hc) / 256, 256, 0, stream>>>(CU, CS, UIN, SIN,
                                                        taus + (size_t)l * Hc, log_dt);
        scan_pass3<<<dim3(NC, Bc), 256, 0, stream>>>(I, taus + (size_t)l * Hc, log_dt,
                                                     UIN, SIN, XB);
    }

    const float qscale = 0.17677669529663687f;   // 1/sqrt(32)
    gemm_bf16_kernel<2><<<dim3(Mc / 128, Hc / 128), 256, 0, stream>>>(XB, WQ, qb, Qb, Mc, Hc, Hc, qscale);
    gemm_bf16_kernel<2><<<dim3(Mc / 128, Hc / 128), 256, 0, stream>>>(XB, WK, kb, Kb, Mc, Hc, Hc, 1.0f);
    gemm_bf16_kernel<2><<<dim3(Mc / 128, Hc / 128), 256, 0, stream>>>(XB, WV, vb, Vb, Mc, Hc, Hc, 1.0f);

    flash_attn_kernel<<<dim3(Sc / 128, Bc * NHc), 256, 0, stream>>>(Qb, Kb, Vb, CTX);

    gemm_bf16_kernel<1><<<dim3(Mc / 128, Hc / 128), 256, 0, stream>>>(CTX, WO, ob, XB2, Mc, Hc, Hc, 1.0f);

    gemm_bf16_kernel<0><<<dim3(Mc / 128, Vc / 128), 256, 0, stream>>>(XB2, WH, head_b,
                                                                      (float*)d_out, Mc, Vc, Hc, 1.0f);
}
